// TAB_ONNX_28424093564931
// MI455X (gfx1250) — matrix operand order repaired and checked by reading, not compiled, not run on hardware
//
#include <hip/hip_runtime.h>
#include <hip/hip_bf16.h>

typedef _Float16 h16;
typedef __attribute__((ext_vector_type(16))) _Float16 v16h;
typedef __attribute__((ext_vector_type(8)))  float    v8f;

#define DIMC   192
#define NHEAD  8
#define HDIM   24
#define GSZ    128
#define MLPD   384
#define BATCH  4
#define HH     128
#define WW     128
#define NPIX   (HH*WW)        // 16384
#define BNTOK  (BATCH*NPIX)   // 65536
#define NGRP   (NPIX/GSZ)     // 128
#define CHUNK  256
#define NCHUNK (NPIX/CHUNK)   // 64

// ---------------- wave32 helpers ----------------
__device__ inline float wred_sum(float v){
#pragma unroll
  for (int m = 16; m >= 1; m >>= 1) v += __shfl_xor(v, m, 32);
  return v;
}

// ---------------- WMMA fragment loaders (CDNA5 f16 16x16x32 layouts) ----------------
union AFragU { v16h v; unsigned u[8]; };
union BFragU { v16h v; uint4 q[2]; };

// A matrix 16x32 f16: lane holds row m=lane&15; VGPR v holds K pair
//   k = (v%4)*2 + (v>=4?16:0) + (lane>=16?8:0)
__device__ inline v16h load_a_frag(const h16* base, int stride, int lane){
  AFragU f;
  const int m = lane & 15;
  const int koff = (lane & 16) ? 8 : 0;
#pragma unroll
  for (int vv = 0; vv < 8; ++vv){
    int k = ((vv & 3) << 1) + ((vv & 4) ? 16 : 0) + koff;
    f.u[vv] = *(const unsigned*)(base + m * stride + k);
  }
  return f.v;
}
// B matrix 32x16 f16: lane holds col n=lane&15; halves j<8 hold k=(lane>=16?8:0)+j, j>=8 hold k=16+(lane>=16?8:0)+(j-8)
// base points to LDS region laid out [n][k] (transposed), so 16 halves are contiguous.
__device__ inline v16h load_b_frag(const h16* base, int stride, int lane){
  BFragU f;
  const int n  = lane & 15;
  const int kb = (lane & 16) >> 1;
  f.q[0] = *(const uint4*)(base + n * stride + kb);
  f.q[1] = *(const uint4*)(base + n * stride + kb + 16);
  return f.v;
}
__device__ inline v8f wmma_f16(v16h a, v16h b, v8f c){
  return __builtin_amdgcn_wmma_f32_16x16x32_f16(false, a, false, b, (short)0, c, false, false);
}

// Async global->LDS staging of one 128x32-half A tile (8KB = 512 x b128 chunks).
// Thread t issues chunks 2t and 2t+1, so wave w stages exactly rows [w*16, w*16+16)
// -- the rows it alone consumes. Completion enforced per-wave via s_wait_asynccnt.
__device__ inline void stage_a_async(const h16* A, int K, h16* dst, int k0, int tid){
#pragma unroll
  for (int i = 0; i < 2; ++i){
    int ch = tid * 2 + i;
    int row = ch >> 2, part = ch & 3;
    unsigned lds = (unsigned)(size_t)(dst + row*32 + part*8);
    unsigned long long g = (unsigned long long)(size_t)(A + (size_t)row*K + k0 + part*8);
    asm volatile("global_load_async_to_lds_b128 %0, %1, off"
                 :: "v"(lds), "v"(g) : "memory");
  }
}
__device__ inline void wait_async_zero(){
  asm volatile("s_wait_asynccnt 0" ::: "memory");
}

// ---------------- weight prep ----------------
__global__ __launch_bounds__(192) void k_wqkv(const float* __restrict__ Wq, const float* __restrict__ Wk,
                                              const float* __restrict__ Wv, h16* __restrict__ out){
  int k = blockIdx.x, j = threadIdx.x;
  out[k*576 + j]       = (h16)Wq[k*192 + j];
  out[k*576 + 192 + j] = (h16)Wk[k*192 + j];
  out[k*576 + 384 + j] = (h16)Wv[k*192 + j];
}
__global__ void k_cvt(const float* __restrict__ in, h16* __restrict__ out, int n){
  int i = blockIdx.x * blockDim.x + threadIdx.x;
  if (i < n) out[i] = (h16)in[i];
}
__global__ __launch_bounds__(192) void k_wpc(const float* __restrict__ Wp, const float* __restrict__ Wc,
                                             h16* __restrict__ out){
  int i = blockIdx.x, j = threadIdx.x;
  float a = 0.f;
  for (int t = 0; t < 192; ++t) a += Wp[i*192 + t] * Wc[t*192 + j];
  out[i*192 + j] = (h16)a;
}
__global__ __launch_bounds__(192) void k_msn(const float* __restrict__ means, h16* __restrict__ msn){
  __shared__ float red[256];
  int m = blockIdx.x, t = threadIdx.x;
  float v = means[m*192 + t];
  red[t] = v * v;
  if (t < 64) red[192 + t] = 0.f;
  __syncthreads();
#pragma unroll
  for (int s = 128; s >= 1; s >>= 1){
    if (t < s) red[t] += red[t + s];
    __syncthreads();
  }
  float inv = 1.f / (sqrtf(red[0]) + 1e-12f);
  msn[m*192 + t] = (h16)(v * inv);
}
// kg padded [h][64][32], vg transposed padded [h][32][64]
__global__ __launch_bounds__(256) void k_kgvg(const float* __restrict__ means, const float* __restrict__ Wkg,
                                              const float* __restrict__ Wvg, h16* __restrict__ kg,
                                              h16* __restrict__ vgT){
  int m = blockIdx.x, t = threadIdx.x;
  if (t < 192){
    float ak = 0.f, av = 0.f;
    for (int c = 0; c < 192; ++c){
      float mv = means[m*192 + c];
      ak += mv * Wkg[c*192 + t];
      av += mv * Wvg[c*192 + t];
    }
    int hdi = t / 24, d = t % 24;
    kg [(hdi*64 + m)*32 + d] = (h16)ak;
    vgT[(hdi*32 + d)*64 + m] = (h16)av;
  } else {
    int j = t - 192;             // 0..63 -> zero padding d=24..31
    int hdi = j >> 3, d = 24 + (j & 7);
    kg [(hdi*64 + m)*32 + d] = (h16)0;
    vgT[(hdi*32 + d)*64 + m] = (h16)0;
  }
}

// ---------------- LN + cluster assignment ----------------
__global__ __launch_bounds__(256) void k_ln1(const float* __restrict__ x, const float* __restrict__ gam,
                                             const float* __restrict__ bet, const h16* __restrict__ msn,
                                             h16* __restrict__ xn, int* __restrict__ belong){
  __shared__ float xT[64][193];
  const int tid = threadIdx.x;
  const size_t nb = (size_t)blockIdx.x * 64;
  const int b  = (int)(nb >> 14);
  const int n0 = (int)(nb & (NPIX - 1));
#pragma unroll
  for (int i = 0; i < 48; ++i){
    int lin = i * 256 + tid;
    int c = lin >> 6, tk = lin & 63;
    xT[tk][c] = x[((size_t)b*192 + c)*NPIX + n0 + tk];
  }
  __syncthreads();
  const int w = tid >> 5, lane = tid & 31;
  for (int q = 0; q < 8; ++q){
    int tk = w*8 + q;
    float v[6]; float s = 0.f;
#pragma unroll
    for (int i = 0; i < 6; ++i){ v[i] = xT[tk][lane + 32*i]; s += v[i]; }
    float mu = wred_sum(s) * (1.f/192.f);
    float vs = 0.f;
#pragma unroll
    for (int i = 0; i < 6; ++i){ float d = v[i] - mu; vs += d*d; }
    float var = wred_sum(vs) * (1.f/192.f);
    float rs = rsqrtf(var + 1e-5f);
    size_t bn = nb + tk;
#pragma unroll
    for (int i = 0; i < 6; ++i){
      int c = lane + 32*i;
      float y = (v[i] - mu) * rs * gam[c] + bet[c];
      xT[tk][c] = y;
      xn[bn*192 + c] = (h16)y;
    }
  }
  __syncthreads();
  for (int q = 0; q < 8; ++q){
    int tk = w*8 + q;
    int m0 = lane, m1 = lane + 32;
    float a0 = 0.f, a1 = 0.f;
    for (int c = 0; c < 192; ++c){
      float y = xT[tk][c];
      a0 += y * (float)msn[m0*192 + c];
      a1 += y * (float)msn[m1*192 + c];
    }
    float bv; int bi;
    if (a1 > a0){ bv = a1; bi = m1; } else { bv = a0; bi = m0; }
#pragma unroll
    for (int mk = 16; mk >= 1; mk >>= 1){
      float ov = __shfl_xor(bv, mk, 32);
      int   oi = __shfl_xor(bi, mk, 32);
      if (ov > bv || (ov == bv && oi < bi)){ bv = ov; bi = oi; }
    }
    if (lane == 0) belong[nb + tk] = bi;
  }
}

// ---------------- stable counting sort ----------------
__global__ __launch_bounds__(256) void k_hist(const int* __restrict__ belong, int* __restrict__ counts){
  __shared__ int cnt[64];
  int ch = blockIdx.x & 63, b = blockIdx.x >> 6, t = threadIdx.x;
  if (t < 64) cnt[t] = 0;
  __syncthreads();
  int bel = belong[(size_t)b*NPIX + ch*CHUNK + t];
  atomicAdd(&cnt[bel], 1);
  __syncthreads();
  if (t < 64) counts[(b*64 + t)*64 + ch] = cnt[t];
}
__global__ __launch_bounds__(64) void k_scan(const int* __restrict__ counts, int* __restrict__ offs){
  __shared__ int sh[64];
  int b = blockIdx.x, cl = threadIdx.x;
  int tot = 0;
  for (int ch = 0; ch < 64; ++ch) tot += counts[(b*64 + cl)*64 + ch];
  sh[cl] = tot;
  __syncthreads();
#pragma unroll
  for (int s = 1; s < 64; s <<= 1){
    int v = (cl >= s) ? sh[cl - s] : 0;
    __syncthreads();
    sh[cl] += v;
    __syncthreads();
  }
  int run = sh[cl] - tot;   // exclusive prefix over clusters
  for (int ch = 0; ch < 64; ++ch){
    int idx = (b*64 + cl)*64 + ch;
    int v = counts[idx];
    offs[idx] = run;
    run += v;
  }
}
__global__ __launch_bounds__(256) void k_rank(const int* __restrict__ belong, const int* __restrict__ offs,
                                              int* __restrict__ inv, int* __restrict__ idxa){
  __shared__ int bl[256];
  int ch = blockIdx.x & 63, b = blockIdx.x >> 6, t = threadIdx.x;
  int n = ch*CHUNK + t;
  int myb = belong[(size_t)b*NPIX + n];
  bl[t] = myb;
  __syncthreads();
  int rank = 0;
  for (int j = 0; j < t; ++j) rank += (bl[j] == myb) ? 1 : 0;
  int p = offs[(b*64 + myb)*64 + ch] + rank;
  inv [(size_t)b*NPIX + n] = p;
  idxa[(size_t)b*NPIX + p] = n;
}

// ---------------- generic WMMA GEMM, 128x64 tile, async-LDS A staging ----------------
// B panel ([64 cols][K], transposed) is preloaded once; the k-loop stages A tiles with
// global_load_async_to_lds_b128 (double-buffered, per-wave completion, no inner barriers).
// mode 0: QKV -> scatter rows to sorted order (e0=inv, e1=q, e2=k, e3=v), fp16 out
// mode 1: proj*conv -> fp32 xt = acc + x residual (e0=x NCHW, e1=xt)
// mode 2: fc1 -> fp16 transposed (b, MLPD, N) (e1=h1t)
// mode 3: fc2 -> fp32 NCHW out = acc + xt residual (e0=xt, e1=out)
__global__ __launch_bounds__(256) void k_gemm(const h16* __restrict__ A, const h16* __restrict__ Bw,
                                              int K, int Ncols, int mode,
                                              const void* e0, void* e1, void* e2, void* e3){
  extern __shared__ __align__(16) h16 gsm[];
  h16* Bs = gsm;                 // [64][K]  (transposed panel)
  h16* As = gsm + 64 * K;        // 2 x [128][32] double buffer
  const int tid = threadIdx.x, lane = tid & 31, w = tid >> 5;
  const size_t rowBase = (size_t)blockIdx.x * 128;
  const int colBase = blockIdx.y * 64;
  const h16* Arow = A + rowBase * (size_t)K;

  // B panel -> LDS transposed [n][k] (coalesced across lanes per k)
  {
    int n = tid & 63, qq = tid >> 6;
    int kpt = K >> 2;
    for (int k = qq * kpt; k < (qq + 1) * kpt; ++k)
      Bs[n*K + k] = Bw[(size_t)k * Ncols + colBase + n];
  }
  // stage tile 0 asynchronously
  stage_a_async(Arow, K, As, 0, tid);

  v8f acc[4];
#pragma unroll
  for (int nt = 0; nt < 4; ++nt)
#pragma unroll
    for (int r = 0; r < 8; ++r) acc[nt][r] = 0.f;

  __syncthreads();               // B panel visible (async A is per-wave, waited below)

  const int nk = K >> 5;
  for (int kt = 0; kt < nk; ++kt){
    wait_async_zero();           // this wave's current A tile has landed
    h16* cur = As + (kt & 1) * (128*32);
    v16h af = load_a_frag(cur + w*16*32, 32, lane);
    if (kt + 1 < nk)             // prefetch next tile into the other buffer
      stage_a_async(Arow, K, As + ((kt + 1) & 1) * (128*32), (kt + 1) * 32, tid);
#pragma unroll
    for (int nt = 0; nt < 4; ++nt){
      v16h bf = load_b_frag(Bs + nt*16*K + kt*32, K, lane);
      acc[nt] = wmma_f16(af, bf, acc[nt]);
    }
  }

  const int nloc  = lane & 15;
  const int mhalf = (lane >> 4) << 3;
  if (mode == 0){
    const int* inv = (const int*)e0;
    h16* qd = (h16*)e1; h16* kd = (h16*)e2; h16* vd = (h16*)e3;
    h16* dst = (colBase < 192) ? qd : (colBase < 384 ? kd : vd);
    int cb = colBase % 192;
#pragma unroll
    for (int nt = 0; nt < 4; ++nt){
      int c = cb + nt*16 + nloc;
#pragma unroll
      for (int r = 0; r < 8; ++r){
        size_t bn = rowBase + w*16 + r + mhalf;
        int b = (int)(bn >> 14);
        int p = inv[bn];
        dst[((size_t)b*NPIX + p) * 192 + c] = (h16)acc[nt][r];
      }
    }
  } else if (mode == 1){
    const float* x = (const float*)e0; float* xt = (float*)e1;
#pragma unroll
    for (int nt = 0; nt < 4; ++nt){
      int c = colBase + nt*16 + nloc;
#pragma unroll
      for (int r = 0; r < 8; ++r){
        size_t bn = rowBase + w*16 + r + mhalf;
        int b = (int)(bn >> 14), n = (int)(bn & (NPIX-1));
        xt[bn*192 + c] = acc[nt][r] + x[((size_t)b*192 + c)*NPIX + n];
      }
    }
  } else if (mode == 2){
    h16* h1t = (h16*)e1;
#pragma unroll
    for (int nt = 0; nt < 4; ++nt){
      int c = colBase + nt*16 + nloc;
#pragma unroll
      for (int r = 0; r < 8; ++r){
        size_t bn = rowBase + w*16 + r + mhalf;
        int b = (int)(bn >> 14), n = (int)(bn & (NPIX-1));
        h1t[((size_t)b*MLPD + c)*NPIX + n] = (h16)acc[nt][r];
      }
    }
  } else {
    const float* xt = (const float*)e0; float* outp = (float*)e1;
#pragma unroll
    for (int nt = 0; nt < 4; ++nt){
      int c = colBase + nt*16 + nloc;
#pragma unroll
      for (int r = 0; r < 8; ++r){
        size_t bn = rowBase + w*16 + r + mhalf;
        int b = (int)(bn >> 14), n = (int)(bn & (NPIX-1));
        outp[((size_t)b*192 + c)*NPIX + n] = acc[nt][r] + xt[bn*192 + c];
      }
    }
  }
}

// ---------------- attention: local windows + global dictionary ----------------
__global__ __launch_bounds__(256) void k_attn(const h16* __restrict__ q_s, const h16* __restrict__ k_s,
                                              const h16* __restrict__ v_s, const h16* __restrict__ kg_p,
                                              const h16* __restrict__ vgT_p, const int* __restrict__ idxa,
                                              h16* __restrict__ attn){
  extern __shared__ __align__(16) h16 smem[];
  h16* qs  = smem;               // 128*32
  h16* ks  = qs  + 128*32;       // 256*32
  h16* vT  = ks  + 256*32;       // 32*256
  h16* kgs = vT  + 32*256;       // 64*32
  h16* vgs = kgs + 64*32;        // 32*64
  h16* pst = vgs + 32*64;        // 8*16*256
  const int g = blockIdx.x, hd = blockIdx.y, b = blockIdx.z;
  const int tid = threadIdx.x, lane = tid & 31, w = tid >> 5;
  const float scale = 0.2041241452f;   // 24^-0.5

  {
    int row = tid >> 1, part = (tid & 1) * 16;
    const h16* src = q_s + ((size_t)b*NPIX + g*GSZ + row)*192 + hd*HDIM;
#pragma unroll
    for (int j = 0; j < 16; ++j){
      int d = part + j;
      qs[row*32 + d] = (d < HDIM) ? src[d] : (h16)0;
    }
  }
  {
    int r = tid;
    int s = g*GSZ + r;
    int kr = (s < NPIX) ? s : (2*NPIX - 1 - s);   // reflect wrap for last group
    const h16* ksrc = k_s + ((size_t)b*NPIX + kr)*192 + hd*HDIM;
    const h16* vsrc = v_s + ((size_t)b*NPIX + kr)*192 + hd*HDIM;
#pragma unroll
    for (int d = 0; d < 32; ++d){
      h16 kv = (d < HDIM) ? ksrc[d] : (h16)0;
      h16 vv = (d < HDIM) ? vsrc[d] : (h16)0;
      ks[r*32 + d]  = kv;
      vT[d*256 + r] = vv;
    }
  }
  ((uint4*)kgs)[tid] = ((const uint4*)(kg_p  + hd*2048))[tid];
  ((uint4*)vgs)[tid] = ((const uint4*)(vgT_p + hd*2048))[tid];
  __syncthreads();

  v16h aq = load_a_frag(qs + w*16*32, 32, lane);
  h16* pw = pst + w*16*256;

  // local scores (K=32 padded head dim -> one WMMA per 16x16 score tile)
  v8f accS[16];
#pragma unroll
  for (int t = 0; t < 16; ++t){
    v8f z = {0.f,0.f,0.f,0.f,0.f,0.f,0.f,0.f};
    v16h bf = load_b_frag(ks + t*16*32, 32, lane);
    accS[t] = wmma_f16(aq, bf, z);
  }
  // softmax over 256 keys (per row), write p (fp16) to LDS staging
#pragma unroll
  for (int r = 0; r < 8; ++r){
    float mx = -1e30f;
#pragma unroll
    for (int t = 0; t < 16; ++t) mx = fmaxf(mx, accS[t][r]);
#pragma unroll
    for (int mk = 8; mk >= 1; mk >>= 1) mx = fmaxf(mx, __shfl_xor(mx, mk, 32));
    float ssum = 0.f;
#pragma unroll
    for (int t = 0; t < 16; ++t){
      float e = __expf((accS[t][r] - mx) * scale);
      accS[t][r] = e;
      ssum += e;
    }
#pragma unroll
    for (int mk = 8; mk >= 1; mk >>= 1) ssum += __shfl_xor(ssum, mk, 32);
    float is = 1.f / ssum;
    int mrow = r + ((lane >> 4) << 3);
#pragma unroll
    for (int t = 0; t < 16; ++t)
      pw[mrow*256 + t*16 + (lane & 15)] = (h16)(accS[t][r] * is);
  }
  __syncthreads();

  // PV local
  v8f accO[2];
#pragma unroll
  for (int nt = 0; nt < 2; ++nt){
    v8f z = {0.f,0.f,0.f,0.f,0.f,0.f,0.f,0.f};
    accO[nt] = z;
  }
#pragma unroll
  for (int kc = 0; kc < 8; ++kc){
    v16h ap = load_a_frag(pw + kc*32, 256, lane);
#pragma unroll
    for (int nt = 0; nt < 2; ++nt){
      v16h bf = load_b_frag(vT + nt*16*256 + kc*32, 256, lane);
      accO[nt] = wmma_f16(ap, bf, accO[nt]);
    }
  }
  __syncthreads();

  // global dictionary attention (64 keys), accumulate into same output (out1+out2)
  v8f accG[4];
#pragma unroll
  for (int t = 0; t < 4; ++t){
    v8f z = {0.f,0.f,0.f,0.f,0.f,0.f,0.f,0.f};
    v16h bf = load_b_frag(kgs + t*16*32, 32, lane);
    accG[t] = wmma_f16(aq, bf, z);
  }
#pragma unroll
  for (int r = 0; r < 8; ++r){
    float mx = -1e30f;
#pragma unroll
    for (int t = 0; t < 4; ++t) mx = fmaxf(mx, accG[t][r]);
#pragma unroll
    for (int mk = 8; mk >= 1; mk >>= 1) mx = fmaxf(mx, __shfl_xor(mx, mk, 32));
    float ssum = 0.f;
#pragma unroll
    for (int t = 0; t < 4; ++t){
      float e = __expf((accG[t][r] - mx) * scale);
      accG[t][r] = e;
      ssum += e;
    }
#pragma unroll
    for (int mk = 8; mk >= 1; mk >>= 1) ssum += __shfl_xor(ssum, mk, 32);
    float is = 1.f / ssum;
    int mrow = r + ((lane >> 4) << 3);
#pragma unroll
    for (int t = 0; t < 4; ++t)
      pw[mrow*256 + t*16 + (lane & 15)] = (h16)(accG[t][r] * is);
  }
  __syncthreads();
#pragma unroll
  for (int kc = 0; kc < 2; ++kc){
    v16h ap = load_a_frag(pw + kc*32, 256, lane);
#pragma unroll
    for (int nt = 0; nt < 2; ++nt){
      v16h bf = load_b_frag(vgs + nt*16*64 + kc*32, 64, lane);
      accO[nt] = wmma_f16(ap, bf, accO[nt]);
    }
  }

  // scatter back to original token order (idx: sorted->orig)
#pragma unroll
  for (int nt = 0; nt < 2; ++nt){
    int d = nt*16 + (lane & 15);
    if (d < HDIM){
#pragma unroll
      for (int r = 0; r < 8; ++r){
        int mrow = r + ((lane >> 4) << 3);
        int p = g*GSZ + w*16 + mrow;
        int n = idxa[(size_t)b*NPIX + p];
        attn[((size_t)b*NPIX + n)*192 + hd*HDIM + d] = (h16)accO[nt][r];
      }
    }
  }
}

// ---------------- post-attention LN ----------------
__global__ __launch_bounds__(256) void k_ln2(const float* __restrict__ xt, const float* __restrict__ gam,
                                             const float* __restrict__ bet, h16* __restrict__ xn2){
  const int tid = threadIdx.x, w = tid >> 5, lane = tid & 31;
  const size_t nb = (size_t)blockIdx.x * 64;
  for (int q = 0; q < 8; ++q){
    size_t bn = nb + w*8 + q;
    float v[6]; float s = 0.f;
#pragma unroll
    for (int i = 0; i < 6; ++i){ v[i] = xt[bn*192 + lane + 32*i]; s += v[i]; }
    float mu = wred_sum(s) * (1.f/192.f);
    float vs = 0.f;
#pragma unroll
    for (int i = 0; i < 6; ++i){ float d = v[i] - mu; vs += d*d; }
    float rs = rsqrtf(wred_sum(vs) * (1.f/192.f) + 1e-5f);
#pragma unroll
    for (int i = 0; i < 6; ++i){
      int c = lane + 32*i;
      xn2[bn*192 + c] = (h16)((v[i] - mu) * rs * gam[c] + bet[c]);
    }
  }
}

// ---------------- depthwise 3x3 + GELU (tanh approx), transpose back ----------------
__global__ __launch_bounds__(128) void k_dwgelu(const h16* __restrict__ h1t, const float* __restrict__ dw,
                                                h16* __restrict__ h2){
  const int xx = threadIdx.x;
  const int y = blockIdx.x, cg = blockIdx.y, b = blockIdx.z;
#pragma unroll
  for (int cc = 0; cc < 8; ++cc){
    int c = cg*8 + cc;
    const h16* base = h1t + ((size_t)b*MLPD + c)*NPIX;
    const float* wv = dw + c*9;
    float acc = 0.f;
#pragma unroll
    for (int dy = -1; dy <= 1; ++dy){
      int yy = y + dy;
      if (yy < 0 || yy >= HH) continue;
#pragma unroll
      for (int dx = -1; dx <= 1; ++dx){
        int xv = xx + dx;
        if (xv < 0 || xv >= WW) continue;
        acc += (float)base[yy*WW + xv] * wv[(dy+1)*3 + (dx+1)];
      }
    }
    float a = acc;
    float gel = 0.5f * a * (1.f + tanhf(0.7978845608f * (a + 0.044715f * a*a*a)));
    h2[((size_t)(b*NPIX + y*WW + xx))*MLPD + c] = (h16)gel;
  }
}

// ---------------- host launcher ----------------
extern "C" void kernel_launch(void* const* d_in, const int* in_sizes, int n_in,
                              void* d_out, int out_size, void* d_ws, size_t ws_size,
                              hipStream_t stream){
  (void)in_sizes; (void)n_in; (void)out_size; (void)ws_size;
  const float* x     = (const float*)d_in[0];
  const float* ln_g  = (const float*)d_in[1];
  const float* ln_b  = (const float*)d_in[2];
  const float* Wq    = (const float*)d_in[3];
  const float* Wk    = (const float*)d_in[4];
  const float* Wv    = (const float*)d_in[5];
  const float* Wproj = (const float*)d_in[6];
  const float* Wkg   = (const float*)d_in[7];
  const float* Wvg   = (const float*)d_in[8];
  const float* means = (const float*)d_in[9];
  const float* Wconv = (const float*)d_in[10];
  const float* mg    = (const float*)d_in[11];
  const float* mb    = (const float*)d_in[12];
  const float* Wfc1  = (const float*)d_in[13];
  const float* dw    = (const float*)d_in[14];
  const float* Wfc2  = (const float*)d_in[15];
  float* out = (float*)d_out;

  char* ws = (char*)d_ws;
  size_t off = 0;
  auto alloc = [&](size_t bytes) -> void* {
    void* p = ws + off;
    off = (off + bytes + 255) & ~(size_t)255;
    return p;
  };
  h16*  xn_h   = (h16*)alloc((size_t)BNTOK*192*2);
  h16*  q_s    = (h16*)alloc((size_t)BNTOK*192*2);
  h16*  k_sb   = (h16*)alloc((size_t)BNTOK*192*2);
  h16*  v_sb   = (h16*)alloc((size_t)BNTOK*192*2);
  h16*  attn_h = (h16*)alloc((size_t)BNTOK*192*2);
  float* xt    = (float*)alloc((size_t)BNTOK*192*4);
  h16*  xn2_h  = (h16*)alloc((size_t)BNTOK*192*2);
  h16*  h1t    = (h16*)alloc((size_t)BNTOK*384*2);
  h16*  h2b    = (h16*)alloc((size_t)BNTOK*384*2);
  h16*  Wqkv_h = (h16*)alloc((size_t)192*576*2);
  h16*  Wpc_h  = (h16*)alloc((size_t)192*192*2);
  h16*  Wfc1_h = (h16*)alloc((size_t)192*384*2);
  h16*  Wfc2_h = (h16*)alloc((size_t)384*192*2);
  h16*  msn_h  = (h16*)alloc((size_t)64*192*2);
  h16*  kg_p   = (h16*)alloc((size_t)8*64*32*2);
  h16*  vgT_p  = (h16*)alloc((size_t)8*32*64*2);
  int*  belong = (int*)alloc((size_t)BNTOK*4);
  int*  invp   = (int*)alloc((size_t)BNTOK*4);
  int*  idxa   = (int*)alloc((size_t)BNTOK*4);
  int*  counts = (int*)alloc((size_t)BATCH*64*64*4);
  int*  offs   = (int*)alloc((size_t)BATCH*64*64*4);

  // weight prep
  k_wqkv<<<192, 192, 0, stream>>>(Wq, Wk, Wv, Wqkv_h);
  k_wpc <<<192, 192, 0, stream>>>(Wproj, Wconv, Wpc_h);
  k_cvt <<<(192*384 + 255)/256, 256, 0, stream>>>(Wfc1, Wfc1_h, 192*384);
  k_cvt <<<(384*192 + 255)/256, 256, 0, stream>>>(Wfc2, Wfc2_h, 384*192);
  k_msn <<<64, 192, 0, stream>>>(means, msn_h);
  k_kgvg<<<64, 256, 0, stream>>>(means, Wkg, Wvg, kg_p, vgT_p);

  // LN + cluster assignment + stable counting sort
  k_ln1 <<<BNTOK/64, 256, 0, stream>>>(x, ln_g, ln_b, msn_h, xn_h, belong);
  k_hist<<<BATCH*NCHUNK, 256, 0, stream>>>(belong, counts);
  k_scan<<<BATCH, 64, 0, stream>>>(counts, offs);
  k_rank<<<BATCH*NCHUNK, 256, 0, stream>>>(belong, offs, invp, idxa);

  // GEMM dynamic LDS: B panel (64*K halves) + A double buffer (2*128*32 halves)
  size_t smem192 = (size_t)(64*192 + 2*128*32) * 2;   // 40 KB
  size_t smem384 = (size_t)(64*384 + 2*128*32) * 2;   // 64 KB
  (void)hipFuncSetAttribute((const void*)k_gemm, hipFuncAttributeMaxDynamicSharedMemorySize,
                            (int)smem384);

  // QKV projections with scatter to sorted order
  k_gemm<<<dim3(BNTOK/128, 9), 256, smem192, stream>>>(xn_h, Wqkv_h, 192, 576, 0,
                                                       invp, q_s, k_sb, v_sb);
  // attention (local windows + global dictionary)
  size_t smemA = (size_t)(128*32 + 256*32 + 32*256 + 64*32 + 32*64 + 8*16*256) * 2;
  (void)hipFuncSetAttribute((const void*)k_attn, hipFuncAttributeMaxDynamicSharedMemorySize,
                            (int)smemA);
  k_attn<<<dim3(NGRP, NHEAD, BATCH), 256, smemA, stream>>>(q_s, k_sb, v_sb, kg_p, vgT_p, idxa, attn_h);

  // proj*conv + residual
  k_gemm<<<dim3(BNTOK/128, 3), 256, smem192, stream>>>(attn_h, Wpc_h, 192, 192, 1,
                                                       (const void*)x, xt, nullptr, nullptr);
  // ConvFFN
  k_ln2 <<<BNTOK/64, 256, 0, stream>>>(xt, mg, mb, xn2_h);
  k_gemm<<<dim3(BNTOK/128, 6), 256, smem192, stream>>>(xn2_h, Wfc1_h, 192, 384, 2,
                                                       nullptr, h1t, nullptr, nullptr);
  k_dwgelu<<<dim3(HH, MLPD/8, BATCH), 128, 0, stream>>>(h1t, dw, h2b);
  k_gemm<<<dim3(BNTOK/128, 3), 256, smem384, stream>>>(h2b, Wfc2_h, 384, 192, 3,
                                                       (const void*)xt, out, nullptr, nullptr);
}
